// SA_Layer_6622839571118
// MI455X (gfx1250) — matrix operand order repaired and checked by reading, not compiled, not run on hardware
//
#include <hip/hip_runtime.h>
#include <hip/hip_bf16.h>

typedef __attribute__((ext_vector_type(16))) __bf16 v16bf;
typedef __attribute__((ext_vector_type(8)))  __bf16 v8bf;
typedef __attribute__((ext_vector_type(8)))  float  v8f;

#define BB 4
#define CC 512
#define NN 4096
#define DD 128

union FragU { v16bf v; v8bf h[2]; };

// A fragment: 16x32 bf16, rows of row-major [rows][ld] buffer.
// ISA layout: lanes 0-15 (row=lane): elems 0-7 = K 0..7, elems 8-15 = K 16..23
//             lanes 16-31 (row=lane-16): elems 0-7 = K 8..15, elems 8-15 = K 24..31
__device__ __forceinline__ v16bf load_fragA(const __bf16* __restrict__ p0, int ld) {
  const int lane = threadIdx.x & 31;
  const int r = lane & 15, h = lane >> 4;
  const __bf16* p = p0 + (size_t)r * ld + h * 8;
  FragU f;
  f.h[0] = *(const v8bf*)(p);
  f.h[1] = *(const v8bf*)(p + 16);
  return f.v;
}

// B fragment: 32x16 bf16, from row-major [cols][ld] ("B-transposed" buffer).
// ISA layout: lanes 0-15 (col=lane): elems 0-7 = K 0..7, elems 8-15 = K 16..23
//             lanes 16-31 (col=lane-16): elems 0-7 = K 8..15, elems 8-15 = K 24..31
__device__ __forceinline__ v16bf load_fragB(const __bf16* __restrict__ p0, int ld) {
  const int lane = threadIdx.x & 31;
  const int r = lane & 15, h = lane >> 4;
  const __bf16* p = p0 + (size_t)r * ld + h * 8;
  FragU f;
  f.h[0] = *(const v8bf*)(p);
  f.h[1] = *(const v8bf*)(p + 16);
  return f.v;
}

// Generic NT GEMM: C[M][Nn] = A[M][K] * B[Nn][K]^T   (bf16 in, fp32 accum)
// Block tile 128(M) x 64(N), 8 waves, wave tile 32x32 (2x2 WMMA 16x16x32 bf16).
template<int OUT_BF16, int HAS_BIAS, int HAS_SCALE>
__global__ __launch_bounds__(256)
void gemm_nt(void* __restrict__ Cout,
             const __bf16* __restrict__ A,
             const __bf16* __restrict__ Bm,
             int Nn, int K,
             long long sA, long long sB, long long sC,
             const float* __restrict__ bias,
             const float* __restrict__ colScale, int sScale)
{
  const int bz = blockIdx.z;
  const __bf16* Ab = A  + (size_t)bz * sA;
  const __bf16* Bb = Bm + (size_t)bz * sB;
  const int wave = threadIdx.x >> 5;
  const int m0 = blockIdx.y * 128 + (wave & 3) * 32;
  const int n0 = blockIdx.x * 64  + (wave >> 2) * 32;

  v8f acc00 = {}, acc01 = {}, acc10 = {}, acc11 = {};
  const __bf16* pa0 = Ab + (size_t)m0 * K;
  const __bf16* pa1 = Ab + (size_t)(m0 + 16) * K;
  const __bf16* pb0 = Bb + (size_t)n0 * K;
  const __bf16* pb1 = Bb + (size_t)(n0 + 16) * K;

  for (int k0 = 0; k0 < K; k0 += 32) {
    v16bf a0 = load_fragA(pa0 + k0, K);
    v16bf a1 = load_fragA(pa1 + k0, K);
    v16bf b0 = load_fragB(pb0 + k0, K);
    v16bf b1 = load_fragB(pb1 + k0, K);
    acc00 = __builtin_amdgcn_wmma_f32_16x16x32_bf16(false, a0, false, b0, (short)0, acc00, false, false);
    acc01 = __builtin_amdgcn_wmma_f32_16x16x32_bf16(false, a0, false, b1, (short)0, acc01, false, false);
    acc10 = __builtin_amdgcn_wmma_f32_16x16x32_bf16(false, a1, false, b0, (short)0, acc10, false, false);
    acc11 = __builtin_amdgcn_wmma_f32_16x16x32_bf16(false, a1, false, b1, (short)0, acc11, false, false);
  }

  // C layout: lanes 0-15: col=lane, elem e -> row m0+e ; lanes 16-31: col=lane-16, row m0+8+e
  const int lane = threadIdx.x & 31;
  const int r = lane & 15, h = lane >> 4;
  v8f accs[2][2] = {{acc00, acc01}, {acc10, acc11}};
  #pragma unroll
  for (int i = 0; i < 2; ++i) {
    #pragma unroll
    for (int j = 0; j < 2; ++j) {
      const int col = n0 + j * 16 + r;
      float scale = 1.0f;
      if (HAS_SCALE) scale = 1.0f / (1e-9f + colScale[(size_t)bz * sScale + col]);
      #pragma unroll
      for (int e = 0; e < 8; ++e) {
        const int row = m0 + i * 16 + h * 8 + e;
        float val = accs[i][j][e];
        if (HAS_BIAS)  val += bias[row];
        if (HAS_SCALE) val *= scale;
        const size_t idx = (size_t)bz * sC + (size_t)row * Nn + col;
        if (OUT_BF16) ((__bf16*)Cout)[idx] = (__bf16)val;
        else          ((float*)Cout)[idx]  = val;
      }
    }
  }
}

__device__ __forceinline__ float wave_sum(float v) {
  #pragma unroll
  for (int o = 16; o > 0; o >>= 1) v += __shfl_xor(v, o, 32);
  return v;
}
__device__ __forceinline__ float wave_max(float v) {
  #pragma unroll
  for (int o = 16; o > 0; o >>= 1) v = fmaxf(v, __shfl_xor(v, o, 32));
  return v;
}

// Per row n of G: rowMax[n], rowRcp[n] = 1/sum(exp(row - max))
__global__ __launch_bounds__(256)
void row_softmax_stats(const float* __restrict__ G, float* __restrict__ rowMax,
                       float* __restrict__ rowRcp)
{
  const int n = blockIdx.x, bz = blockIdx.y;
  const float* row = G + ((size_t)bz * NN + n) * NN;
  const int t = threadIdx.x;
  float v[16];
  float mx = -3.0e38f;
  #pragma unroll
  for (int i = 0; i < 16; ++i) { v[i] = row[t + i * 256]; mx = fmaxf(mx, v[i]); }
  mx = wave_max(mx);
  __shared__ float s[8];
  if ((t & 31) == 0) s[t >> 5] = mx;
  __syncthreads();
  mx = s[0];
  #pragma unroll
  for (int w = 1; w < 8; ++w) mx = fmaxf(mx, s[w]);
  __syncthreads();
  float sum = 0.f;
  #pragma unroll
  for (int i = 0; i < 16; ++i) sum += __expf(v[i] - mx);
  sum = wave_sum(sum);
  if ((t & 31) == 0) s[t >> 5] = sum;
  __syncthreads();
  if (t == 0) {
    float tot = 0.f;
    #pragma unroll
    for (int w = 0; w < 8; ++w) tot += s[w];
    rowMax[(size_t)bz * NN + n] = mx;
    rowRcp[(size_t)bz * NN + n] = 1.0f / tot;
  }
}

// Using G symmetry: W[m][n] = attn^T[m][n]*colS[m] = exp(G[m][n]-rm[n])*rz[n]  (bf16)
// colS[m] = sum_n W[m][n]  (query-dim L1 sum of the row-softmaxed attn)
__global__ __launch_bounds__(256)
void attn_weights_colsum(const float* __restrict__ G, const float* __restrict__ rowMax,
                         const float* __restrict__ rowRcp,
                         __bf16* __restrict__ W, float* __restrict__ colSum)
{
  const int m = blockIdx.x, bz = blockIdx.y;
  const float* row = G + ((size_t)bz * NN + m) * NN;
  const float* rmv = rowMax + (size_t)bz * NN;
  const float* rzv = rowRcp + (size_t)bz * NN;
  __bf16* wrow = W + ((size_t)bz * NN + m) * NN;
  float s = 0.f;
  for (int i = threadIdx.x; i < NN; i += 256) {
    float w = __expf(row[i] - rmv[i]) * rzv[i];
    wrow[i] = (__bf16)w;
    s += w;
  }
  s = wave_sum(s);
  __shared__ float sb[8];
  if ((threadIdx.x & 31) == 0) sb[threadIdx.x >> 5] = s;
  __syncthreads();
  if (threadIdx.x == 0) {
    float tot = 0.f;
    #pragma unroll
    for (int w = 0; w < 8; ++w) tot += sb[w];
    colSum[(size_t)bz * NN + m] = tot;
  }
}

// BatchNorm training-mode stats per channel d over (B, N)
__global__ __launch_bounds__(256)
void bn_stats(const float* __restrict__ xr, float* __restrict__ meanOut,
              float* __restrict__ rstdOut)
{
  const int d = blockIdx.x;
  float s = 0.f, ss = 0.f;
  for (int bz = 0; bz < BB; ++bz) {
    const float* p = xr + ((size_t)bz * DD + d) * NN;
    for (int i = threadIdx.x; i < NN; i += 256) { float v = p[i]; s += v; ss += v * v; }
  }
  s = wave_sum(s); ss = wave_sum(ss);
  __shared__ float sb[8], sb2[8];
  if ((threadIdx.x & 31) == 0) { sb[threadIdx.x >> 5] = s; sb2[threadIdx.x >> 5] = ss; }
  __syncthreads();
  if (threadIdx.x == 0) {
    float ts = 0.f, tss = 0.f;
    #pragma unroll
    for (int w = 0; w < 8; ++w) { ts += sb[w]; tss += sb2[w]; }
    const float cnt = (float)(BB * NN);
    const float mean = ts / cnt;
    const float var = tss / cnt - mean * mean;
    meanOut[d] = mean;
    rstdOut[d] = rsqrtf(var + 1e-5f);
  }
}

__global__ __launch_bounds__(256)
void bn_relu(const float* __restrict__ xr, const float* __restrict__ mean,
             const float* __restrict__ rstd, const float* __restrict__ gamma,
             const float* __restrict__ beta, float* __restrict__ out)
{
  const size_t i = (size_t)blockIdx.x * 256 + threadIdx.x;
  const int d = (int)((i / NN) % DD);
  const float v = (xr[i] - mean[d]) * rstd[d] * gamma[d] + beta[d];
  out[i] = fmaxf(v, 0.f);
}

// x [B][C][N] fp32 -> xT [B][N][C] bf16 (LDS-tiled transpose + convert)
__global__ __launch_bounds__(256)
void transpose_to_bf16(const float* __restrict__ x, __bf16* __restrict__ xT)
{
  __shared__ __bf16 tile[32][33];
  const int bz = blockIdx.z;
  const float* xp = x + (size_t)bz * CC * NN;
  __bf16* xtp = xT + (size_t)bz * NN * CC;
  const int n0 = blockIdx.x * 32, c0 = blockIdx.y * 32;
  const int tx = threadIdx.x & 31, ty = threadIdx.x >> 5;
  #pragma unroll
  for (int i = 0; i < 32; i += 8)
    tile[ty + i][tx] = (__bf16)xp[(size_t)(c0 + ty + i) * NN + n0 + tx];
  __syncthreads();
  #pragma unroll
  for (int i = 0; i < 32; i += 8)
    xtp[(size_t)(n0 + ty + i) * CC + c0 + tx] = tile[tx][ty + i];
}

__global__ __launch_bounds__(256)
void convert_w(const float* __restrict__ wq, const float* __restrict__ wv,
               __bf16* __restrict__ wqb, __bf16* __restrict__ wvb)
{
  const int i = blockIdx.x * 256 + threadIdx.x;
  if (i < DD * CC) { wqb[i] = (__bf16)wq[i]; wvb[i] = (__bf16)wv[i]; }
}

extern "C" void kernel_launch(void* const* d_in, const int* in_sizes, int n_in,
                              void* d_out, int out_size, void* d_ws, size_t ws_size,
                              hipStream_t stream) {
  (void)in_sizes; (void)n_in; (void)out_size; (void)ws_size;
  const float* x     = (const float*)d_in[0];
  const float* w_qk  = (const float*)d_in[1];
  const float* w_v   = (const float*)d_in[2];
  const float* b_v   = (const float*)d_in[3];
  const float* gamma = (const float*)d_in[4];
  const float* beta  = (const float*)d_in[5];
  float* out = (float*)d_out;

  char* ws = (char*)d_ws;
  size_t off = 0;
  auto carve = [&](size_t bytes) -> char* {
    char* p = ws + off;
    off += (bytes + 255) & ~(size_t)255;
    return p;
  };
  __bf16* xT     = (__bf16*)carve((size_t)BB * NN * CC * 2);
  __bf16* wqb    = (__bf16*)carve((size_t)DD * CC * 2);
  __bf16* wvb    = (__bf16*)carve((size_t)DD * CC * 2);
  __bf16* qkT    = (__bf16*)carve((size_t)BB * NN * DD * 2);
  __bf16* vB     = (__bf16*)carve((size_t)BB * DD * NN * 2);
  float*  G      = (float*) carve((size_t)BB * NN * NN * 4);
  __bf16* W      = (__bf16*)carve((size_t)BB * NN * NN * 2);
  float*  rowMax = (float*) carve((size_t)BB * NN * 4);
  float*  rowRcp = (float*) carve((size_t)BB * NN * 4);
  float*  colSum = (float*) carve((size_t)BB * NN * 4);
  float*  xr     = (float*) carve((size_t)BB * DD * NN * 4);
  float*  meanB  = (float*) carve((size_t)DD * 4);
  float*  rstdB  = (float*) carve((size_t)DD * 4);

  // 1) precision conversion / transpose
  convert_w<<<dim3((DD * CC) / 256), 256, 0, stream>>>(w_qk, w_v, wqb, wvb);
  transpose_to_bf16<<<dim3(NN / 32, CC / 32, BB), 256, 0, stream>>>(x, xT);

  // 2) qkT[b][n][d] = xT[b] (N x C) * w_qk^T    (shared q/k projection)
  gemm_nt<1, 0, 0><<<dim3(DD / 64, NN / 128, BB), 256, 0, stream>>>(
      qkT, xT, wqb, DD, CC,
      (long long)NN * CC, 0LL, (long long)NN * DD, nullptr, nullptr, 0);

  // 3) v[b][d][n] = w_v (D x C) * x[b] + b_v
  gemm_nt<1, 1, 0><<<dim3(NN / 64, DD / 128, BB), 256, 0, stream>>>(
      vB, wvb, xT, NN, CC,
      0LL, (long long)NN * CC, (long long)DD * NN, b_v, nullptr, 0);

  // 4) G[b] = qkT * qkT^T   (symmetric energy Gram matrix, fp32)
  gemm_nt<0, 0, 0><<<dim3(NN / 64, NN / 128, BB), 256, 0, stream>>>(
      G, qkT, qkT, NN, DD,
      (long long)NN * DD, (long long)NN * DD, (long long)NN * NN, nullptr, nullptr, 0);

  // 5) row softmax stats, then bf16 attn^T weights + query-dim sums
  row_softmax_stats<<<dim3(NN, BB), 256, 0, stream>>>(G, rowMax, rowRcp);
  attn_weights_colsum<<<dim3(NN, BB), 256, 0, stream>>>(G, rowMax, rowRcp, W, colSum);

  // 6) x_r[b][d][m] = (1/(1e-9+colS[m])) * sum_n v[d][n] * W[m][n]
  gemm_nt<0, 0, 1><<<dim3(NN / 64, DD / 128, BB), 256, 0, stream>>>(
      xr, vB, W, NN, NN,
      (long long)DD * NN, (long long)NN * NN, (long long)DD * NN, nullptr, colSum, NN);

  // 7) BatchNorm (batch stats) + ReLU
  bn_stats<<<dim3(DD), 256, 0, stream>>>(xr, meanB, rstdB);
  bn_relu<<<dim3((BB * DD * NN) / 256), 256, 0, stream>>>(xr, meanB, rstdB, gamma, beta, out);
}
